// M_43026982371413
// MI455X (gfx1250) — matrix operand order repaired and checked by reading, not compiled, not run on hardware
//
#include <hip/hip_runtime.h>
#include <stdint.h>

typedef __attribute__((ext_vector_type(16))) _Float16 v16h;
typedef __attribute__((ext_vector_type(8)))  _Float16 v8h;
typedef __attribute__((ext_vector_type(4)))  _Float16 v4h;
typedef __attribute__((ext_vector_type(8)))  float    v8f;
typedef __attribute__((ext_vector_type(4)))  float    f32x4;
typedef __attribute__((ext_vector_type(4)))  int      i32x4;

#define M_TOTAL 8192   // BATCH * SEQ
#define N_TOTAL 4096   // OUT_DIM
#define K_TOTAL 4096   // IN_DIM

// =====================================================================
// Prepack kernels: one-time f32/i32 -> f16 conversion into workspace.
// =====================================================================
__global__ __launch_bounds__(256)
void cvt_f32_to_f16(const float* __restrict__ in, _Float16* __restrict__ out) {
    const size_t i = ((size_t)blockIdx.x * 256 + threadIdx.x) * 8;
    f32x4 a = *(const f32x4*)(in + i);
    f32x4 b = *(const f32x4*)(in + i + 4);
    v8h h;
#pragma unroll
    for (int c = 0; c < 4; ++c) { h[c] = (_Float16)a[c]; h[4 + c] = (_Float16)b[c]; }
    *(v8h*)(out + i) = h;
}

__global__ __launch_bounds__(256)
void cvt_i32_to_f16(const int* __restrict__ in, _Float16* __restrict__ out) {
    const size_t i = ((size_t)blockIdx.x * 256 + threadIdx.x) * 8;
    i32x4 a = *(const i32x4*)(in + i);
    i32x4 b = *(const i32x4*)(in + i + 4);
    v8h h;
#pragma unroll
    for (int c = 0; c < 4; ++c) { h[c] = (_Float16)a[c]; h[4 + c] = (_Float16)b[c]; }
    *(v8h*)(out + i) = h;   // |w| <= 128: exact in f16
}

// =====================================================================
// Main GEMM on prepacked f16: block tile 128(M) x 256(N), wave tile 64x64,
// BK=32. 4 LDS buffers, 3 async-copy stages in flight
// (GLOBAL_LOAD_ASYNC_TO_LDS_B128 + S_WAIT_ASYNCCNT pipeline).
// =====================================================================
#define BM 128
#define BN 256
#define BK 32
#define BKP 40                        // padded row stride (halves): 80B, 16B-aligned
#define NBUF 4
#define AS_BUF_BYTES (BM * BKP * 2)   // 10240
#define BS_BUF_BYTES (BN * BKP * 2)   // 20480

__global__ __launch_bounds__(256, 1)
void qgemm_wmma_f16(const _Float16* __restrict__ A,   // [M][K] f16
                    const _Float16* __restrict__ B,   // [N][K] f16
                    const float* __restrict__ scale,
                    float* __restrict__ out)
{
    __shared__ _Float16 As[NBUF][BM * BKP];   // 40 KB
    __shared__ _Float16 Bs[NBUF][BN * BKP];   // 80 KB

    const int tid  = threadIdx.x;
    const int lane = tid & 31;
    const int wave = tid >> 5;
    const int half = lane >> 4;
    const int r    = lane & 15;

    const int wave_m = wave & 1;   // 2 slabs of 64 rows
    const int wave_n = wave >> 1;  // 4 slabs of 64 cols

    const int m_blk = blockIdx.y * BM;
    const int n_blk = blockIdx.x * BN;

    v8f acc[4][4];
#pragma unroll
    for (int mt = 0; mt < 4; ++mt)
#pragma unroll
        for (int nt = 0; nt < 4; ++nt)
            acc[mt][nt] = v8f{};

    // ---- async-copy chunk mapping: 16B chunks, 4 per 64B row ----
    const int crow = tid >> 2;              // 0..63
    const int cofs = (tid & 3) * 16;        // byte offset within row

    // low 32 bits of a flat LDS pointer == LDS byte address
    const uint32_t asBase = (uint32_t)(uintptr_t)(&As[0][0]);
    const uint32_t bsBase = (uint32_t)(uintptr_t)(&Bs[0][0]);

    uint32_t voffA[2], voffB[4];     // global byte offsets (32-bit, saddr form)
    uint32_t ldsA0[2], ldsB0[4];     // buffer-0 LDS byte addresses
#pragma unroll
    for (int i = 0; i < 2; ++i) {
        const int row = crow + i * 64;                       // 0..127
        voffA[i] = (uint32_t)((m_blk + row) * K_TOTAL * 2 + cofs);
        ldsA0[i] = asBase + row * (BKP * 2) + cofs;
    }
#pragma unroll
    for (int i = 0; i < 4; ++i) {
        const int row = crow + i * 64;                       // 0..255
        voffB[i] = (uint32_t)((n_blk + row) * K_TOTAL * 2 + cofs);
        ldsB0[i] = bsBase + row * (BKP * 2) + cofs;
    }

    const int NT = K_TOTAL / BK;  // 128 (multiple of 4)

    // issue 6 async global->LDS b128 copies for K-step kt into buffer buf.
    // buf is always a compile-time literal at call sites -> inline-const adds.
    auto issue_async = [&](int kt, int buf) {
        const char* aP = (const char*)A + (size_t)kt * (BK * 2);  // uniform SGPR base
        const char* bP = (const char*)B + (size_t)kt * (BK * 2);
        const uint32_t aoff = (uint32_t)buf * AS_BUF_BYTES;
        const uint32_t boff = (uint32_t)buf * BS_BUF_BYTES;
#pragma unroll
        for (int i = 0; i < 2; ++i)
            asm volatile("global_load_async_to_lds_b128 %0, %1, %2"
                         :: "v"(ldsA0[i] + aoff), "v"(voffA[i]), "s"(aP) : "memory");
#pragma unroll
        for (int i = 0; i < 4; ++i)
            asm volatile("global_load_async_to_lds_b128 %0, %1, %2"
                         :: "v"(ldsB0[i] + boff), "v"(voffB[i]), "s"(bP) : "memory");
    };

    auto compute = [&](int buf) {
        v16h bf[4];
#pragma unroll
        for (int nt = 0; nt < 4; ++nt) {
            // B 32x16 layout: N across lanes; lanes<16 -> K {0..7,16..23}; lanes>=16 -> K {8..15,24..31}
            const int row = wave_n * 64 + nt * 16 + r;
            v8h lo = *(const v8h*)&Bs[buf][row * BKP + half * 8];
            v8h hi = *(const v8h*)&Bs[buf][row * BKP + 16 + half * 8];
#pragma unroll
            for (int e = 0; e < 8; ++e) { bf[nt][e] = lo[e]; bf[nt][8 + e] = hi[e]; }
        }
#pragma unroll
        for (int mt = 0; mt < 4; ++mt) {
            // A 16x32 layout: lanes<16 -> K {0..7,16..23}; lanes>=16 -> K {8..15,24..31}
            const int row = wave_m * 64 + mt * 16 + r;
            v8h lo = *(const v8h*)&As[buf][row * BKP + half * 8];
            v8h hi = *(const v8h*)&As[buf][row * BKP + 16 + half * 8];
            v16h af;
#pragma unroll
            for (int e = 0; e < 8; ++e) { af[e] = lo[e]; af[8 + e] = hi[e]; }
#pragma unroll
            for (int nt = 0; nt < 4; ++nt)
                acc[mt][nt] = __builtin_amdgcn_wmma_f32_16x16x32_f16(
                    false, af, false, bf[nt], (short)0, acc[mt][nt], false, false);
        }
    };

    // ---- prologue: 3 stages in flight ----
    issue_async(0, 0);
    issue_async(1, 1);
    issue_async(2, 2);

    // steady state: stages 0..NT-5; each stage k: wait(k done, 12 newer allowed),
    // barrier, issue(k+3) into buffer (k+3)&3, compute buffer k&3.
    for (int kt = 0; kt < NT - 4; kt += 4) {
#pragma unroll
        for (int j = 0; j < 4; ++j) {
            asm volatile("s_wait_asynccnt 0xc" ::: "memory");
            __syncthreads();
            issue_async(kt + j + 3, (j + 3) & 3);
            compute(j);
        }
    }

    // ---- tail: stages NT-4 .. NT-1 ----
    asm volatile("s_wait_asynccnt 0xc" ::: "memory");
    __syncthreads();
    issue_async(NT - 1, 3);
    compute(0);

    asm volatile("s_wait_asynccnt 0xc" ::: "memory");
    __syncthreads();
    compute(1);

    asm volatile("s_wait_asynccnt 0x6" ::: "memory");
    __syncthreads();
    compute(2);

    asm volatile("s_wait_asynccnt 0x0" ::: "memory");
    __syncthreads();
    compute(3);

    // epilogue: C/D layout -> VGPR e: M = e + 8*half, N = lane%16
#pragma unroll
    for (int nt = 0; nt < 4; ++nt) {
        const int n  = n_blk + wave_n * 64 + nt * 16 + r;
        const float sc = scale[n];
#pragma unroll
        for (int mt = 0; mt < 4; ++mt) {
            const int m0 = m_blk + wave_m * 64 + mt * 16 + half * 8;
#pragma unroll
            for (int e = 0; e < 8; ++e)
                out[(size_t)(m0 + e) * N_TOTAL + n] = acc[mt][nt][e] * sc;
        }
    }
}

// =====================================================================
// Fallback fused kernel (round-1): used if ws_size is too small.
// =====================================================================
#define FBM 128
#define FBN 128

__global__ __launch_bounds__(256)
void qlinear_wmma_fused(const float* __restrict__ x,
                        const int*   __restrict__ w,
                        const float* __restrict__ scale,
                        float* __restrict__ out)
{
    __shared__ _Float16 As[2][FBM * BKP];
    __shared__ _Float16 Bs[2][FBN * BKP];

    const int tid  = threadIdx.x;
    const int lane = tid & 31;
    const int wave = tid >> 5;
    const int half = lane >> 4;
    const int r    = lane & 15;

    const int wave_m = wave & 3;
    const int wave_n = wave >> 2;

    const int m_blk = blockIdx.y * FBM;
    const int n_blk = blockIdx.x * FBN;

    v8f acc[2][4];
#pragma unroll
    for (int mt = 0; mt < 2; ++mt)
#pragma unroll
        for (int nt = 0; nt < 4; ++nt)
            acc[mt][nt] = v8f{};

    f32x4 a_reg[4];
    i32x4 b_reg[4];

    const int NT = K_TOTAL / BK;

    auto load_regs = [&](int kt) {
        const int kb = kt * BK;
#pragma unroll
        for (int i = 0; i < 4; ++i) {
            const int g   = tid + i * 256;
            const int row = g >> 3;
            const int kc  = (g & 7) << 2;
            a_reg[i] = *(const f32x4*)(x + (size_t)(m_blk + row) * K_TOTAL + kb + kc);
            b_reg[i] = *(const i32x4*)(w + (size_t)(n_blk + row) * K_TOTAL + kb + kc);
        }
    };

    auto store_lds = [&](int buf) {
#pragma unroll
        for (int i = 0; i < 4; ++i) {
            const int g   = tid + i * 256;
            const int row = g >> 3;
            const int kc  = (g & 7) << 2;
            v4h av, bv;
#pragma unroll
            for (int c = 0; c < 4; ++c) {
                av[c] = (_Float16)a_reg[i][c];
                bv[c] = (_Float16)b_reg[i][c];
            }
            *(v4h*)&As[buf][row * BKP + kc] = av;
            *(v4h*)&Bs[buf][row * BKP + kc] = bv;
        }
    };

    auto compute = [&](int buf) {
        v16h af[2];
        v16h bf[4];
#pragma unroll
        for (int mt = 0; mt < 2; ++mt) {
            const int row = wave_m * 32 + mt * 16 + r;
            v8h lo = *(const v8h*)&As[buf][row * BKP + half * 8];
            v8h hi = *(const v8h*)&As[buf][row * BKP + 16 + half * 8];
#pragma unroll
            for (int e = 0; e < 8; ++e) { af[mt][e] = lo[e]; af[mt][8 + e] = hi[e]; }
        }
#pragma unroll
        for (int nt = 0; nt < 4; ++nt) {
            const int row = wave_n * 64 + nt * 16 + r;
            v8h lo = *(const v8h*)&Bs[buf][row * BKP + half * 8];
            v8h hi = *(const v8h*)&Bs[buf][row * BKP + 16 + half * 8];
#pragma unroll
            for (int e = 0; e < 8; ++e) { bf[nt][e] = lo[e]; bf[nt][8 + e] = hi[e]; }
        }
#pragma unroll
        for (int mt = 0; mt < 2; ++mt)
#pragma unroll
            for (int nt = 0; nt < 4; ++nt)
                acc[mt][nt] = __builtin_amdgcn_wmma_f32_16x16x32_f16(
                    false, af[mt], false, bf[nt], (short)0, acc[mt][nt], false, false);
    };

    load_regs(0);
    store_lds(0);
    __syncthreads();

    for (int kt = 0; kt < NT; ++kt) {
        const int cur = kt & 1;
        if (kt + 1 < NT) load_regs(kt + 1);
        compute(cur);
        if (kt + 1 < NT) store_lds(1 - cur);
        __syncthreads();
    }

#pragma unroll
    for (int nt = 0; nt < 4; ++nt) {
        const int n  = n_blk + wave_n * 64 + nt * 16 + r;
        const float sc = scale[n];
#pragma unroll
        for (int mt = 0; mt < 2; ++mt) {
            const int m0 = m_blk + wave_m * 32 + mt * 16 + half * 8;
#pragma unroll
            for (int e = 0; e < 8; ++e)
                out[(size_t)(m0 + e) * N_TOTAL + n] = acc[mt][nt][e] * sc;
        }
    }
}

// =====================================================================
extern "C" void kernel_launch(void* const* d_in, const int* in_sizes, int n_in,
                              void* d_out, int out_size, void* d_ws, size_t ws_size,
                              hipStream_t stream) {
    (void)in_sizes; (void)n_in; (void)out_size;
    const float* x     = (const float*)d_in[0];
    const int*   w     = (const int*)d_in[1];
    const float* scale = (const float*)d_in[2];
    float*       out   = (float*)d_out;

    const size_t XH_BYTES = (size_t)M_TOTAL * K_TOTAL * sizeof(_Float16);  // 64 MiB
    const size_t WH_BYTES = (size_t)N_TOTAL * K_TOTAL * sizeof(_Float16);  // 32 MiB

    if (ws_size >= XH_BYTES + WH_BYTES) {
        _Float16* xh = (_Float16*)d_ws;
        _Float16* wh = (_Float16*)((char*)d_ws + XH_BYTES);

        cvt_f32_to_f16<<<(M_TOTAL * (size_t)K_TOTAL) / 2048, 256, 0, stream>>>(x, xh);
        cvt_i32_to_f16<<<(N_TOTAL * (size_t)K_TOTAL) / 2048, 256, 0, stream>>>(w, wh);

        dim3 grid(N_TOTAL / BN, M_TOTAL / BM);   // (16, 64)
        qgemm_wmma_f16<<<grid, 256, 0, stream>>>(xh, wh, scale, out);
    } else {
        dim3 grid(N_TOTAL / FBN, M_TOTAL / FBM); // (32, 64)
        qlinear_wmma_fused<<<grid, 256, 0, stream>>>(x, w, scale, out);
    }
}
